// DynamicGraphGenerator_49417893708422
// MI455X (gfx1250) — matrix operand order repaired and checked by reading, not compiled, not run on hardware
//
#include <hip/hip_runtime.h>
#include <hip/hip_bf16.h>
#include <math.h>
#include <stdint.h>

// ---- problem constants (from reference) ----
#define BB 32
#define TT 96
#define NN 1024
#define ED 64
#define FFT_DIM 49
#define TOPK 30

typedef __attribute__((ext_vector_type(16))) __bf16 v16bf;
typedef __attribute__((ext_vector_type(8)))  float  v8f;

// ---- workspace layout (bytes, 256-aligned) ----
#define OFF_XN1   ((size_t)0)            //  32*49*1024*4  = 6,422,528
#define OFF_XEDW  ((size_t)6422528)      //  32*1024*256*2 = 16,777,216
#define OFF_X1    ((size_t)23199744)     //  32*1024*64*4  = 8,388,608
#define OFF_STATS ((size_t)31588352)     //  256
#define OFF_X1SWZ ((size_t)31588608)     //  32*65536*2    = 4,194,304 (B-swizzled bf16)

// =====================================================================
// Kernel 1: DFT magnitude (rfft, T=96 -> 49 bins) + first L2 norm (axis=k)
// =====================================================================
__global__ void k_dft(const float* __restrict__ x, float* __restrict__ xn1) {
    __shared__ float ct[TT], st[TT];
    int tid = threadIdx.x;
    if (tid < TT) {
        float ang = -2.0f * 3.14159265358979323846f * (float)tid / (float)TT;
        float s, c;
        __sincosf(ang, &s, &c);
        ct[tid] = c; st[tid] = s;
    }
    __syncthreads();

    int gid = blockIdx.x * blockDim.x + tid;     // 0 .. B*N-1
    int b = gid / NN, n = gid % NN;

    float xr[TT];
    #pragma unroll
    for (int t = 0; t < TT; ++t)
        xr[t] = x[((size_t)b * TT + t) * NN + n];

    float sumsq = 0.0f;
    for (int k = 0; k < FFT_DIM; ++k) {
        float re = 0.0f, im = 0.0f;
        int a = 0;
        #pragma unroll
        for (int t = 0; t < TT; ++t) {
            re = fmaf(xr[t], ct[a], re);
            im = fmaf(xr[t], st[a], im);
            a += k; if (a >= TT) a -= TT;
        }
        sumsq += re * re + im * im;
    }
    float inv = 1.0f / fmaxf(sqrtf(sumsq), 1e-12f);

    for (int k = 0; k < FFT_DIM; ++k) {
        float re = 0.0f, im = 0.0f;
        int a = 0;
        #pragma unroll
        for (int t = 0; t < TT; ++t) {
            re = fmaf(xr[t], ct[a], re);
            im = fmaf(xr[t], st[a], im);
            a += k; if (a >= TT) a -= TT;
        }
        float mag = sqrtf(re * re + im * im);
        xn1[((size_t)b * FFT_DIM + k) * NN + n] = mag * inv;
    }
}

// =====================================================================
// Kernel 2: second L2 norm (axis=n)
// =====================================================================
__global__ void k_norm2(float* __restrict__ xn1) {
    float* p = xn1 + (size_t)blockIdx.x * NN;
    __shared__ float red[256];
    float s = 0.0f;
    for (int i = threadIdx.x; i < NN; i += 256) { float v = p[i]; s += v * v; }
    red[threadIdx.x] = s;
    __syncthreads();
    for (int w = 128; w; w >>= 1) {
        if (threadIdx.x < w) red[threadIdx.x] += red[threadIdx.x + w];
        __syncthreads();
    }
    float inv = 1.0f / fmaxf(sqrtf(red[0]), 1e-12f);
    for (int i = threadIdx.x; i < NN; i += 256) p[i] *= inv;
}

// =====================================================================
// Kernel 3: xf = xn1 @ Wx (K=49) and build bf16 xedw = cat(xf,E,T_D,D_W)
// =====================================================================
__global__ void k_xedw(const float* __restrict__ xn1, const float* __restrict__ E,
                       const float* __restrict__ T_D, const float* __restrict__ D_W,
                       const float* __restrict__ Wx, __bf16* __restrict__ xedw) {
    __shared__ float wsh[FFT_DIM * ED];
    for (int i = threadIdx.x; i < FFT_DIM * ED; i += 256) wsh[i] = Wx[i];
    __syncthreads();

    int d  = threadIdx.x & 63;
    int nl = threadIdx.x >> 6;
    int b  = blockIdx.x / (NN / 4);
    int n  = (blockIdx.x % (NN / 4)) * 4 + nl;

    float acc = 0.0f;
    for (int k = 0; k < FFT_DIM; ++k)
        acc = fmaf(xn1[((size_t)b * FFT_DIM + k) * NN + n], wsh[k * ED + d], acc);

    size_t base = ((size_t)b * NN + n) * 256;
    size_t bn   = ((size_t)b * NN + n) * ED;
    xedw[base + d]        = (__bf16)acc;
    xedw[base + 64 + d]   = (__bf16)E[(size_t)n * ED + d];
    xedw[base + 128 + d]  = (__bf16)T_D[bn + d];
    xedw[base + 192 + d]  = (__bf16)D_W[bn + d];
}

// =====================================================================
// WMMA fragment swizzle helpers (16-bit A 16x32 / B 32x16 / f32 C 16x16):
// LDS tiles are stored pre-swizzled as [tile][kk][lane(32)][e(16 contig bf16)]
// so a fragment load is one 32B per-lane load (2x ds_load_b128).
//   A source (m, kr):  lane = (kr>>3 & 1)*16 + (m&15), e = (kr&7)|((kr>>4)<<3)
//   B source (kr, col): as A, col for m; dew(A)/x1swz(B) use lane = (kr>>4)*16 + m|col, e = kr&15
// =====================================================================

// =====================================================================
// Kernel 4: per-node bmm  x1[b,n,:] = relu(xedw[b,n,:] @ Wd[n])
// [32,256] x [256,64]; 8 waves -> 8 tiles; K-loop 8x v_wmma bf16
// Also emits x1 in bf16, B-swizzled for kernel 6 (x1swz).
// =====================================================================
__global__ void __launch_bounds__(256) k_node_mm(const __bf16* __restrict__ xedw,
                                                 const float* __restrict__ Wd,
                                                 float* __restrict__ x1,
                                                 __bf16* __restrict__ x1swz) {
    int n = blockIdx.x;
    __shared__ __align__(32) __bf16 As[2 * 8 * 32 * 16];   // 16 KB, [mt][kk][lane][e]
    __shared__ __align__(32) __bf16 Bs[4 * 8 * 32 * 16];   // 32 KB, [nt][kk][lane][e]

    const float* wd = Wd + (size_t)n * 256 * 64;
    if (n + 1 < NN)  // prefetch next node's weights (global_prefetch_b8)
        __builtin_prefetch(Wd + (size_t)(n + 1) * 256 * 64 + threadIdx.x * 64, 0, 1);

    // fill B swizzled: source (k, col) of Wd[n]
    for (int i = threadIdx.x; i < 256 * 64; i += 256) {
        int k = i >> 6, col = i & 63;
        int kk = k >> 5, kr = k & 31;
        int lane_ = ((kr >> 3) & 1) * 16 + (col & 15);
        int e = (kr & 7) | ((kr >> 4) << 3), nt = col >> 4;
        Bs[(((nt << 3) + kk) * 32 + lane_) * 16 + e] = (__bf16)wd[i];
    }
    // fill A swizzled: source (m, kcol) of xedw[:, n, :]
    for (int i = threadIdx.x; i < 32 * 256; i += 256) {
        int m = i >> 8, kcol = i & 255;
        int kk = kcol >> 5, kr = kcol & 31;
        int lane_ = ((kr >> 3) & 1) * 16 + (m & 15);
        int e = (kr & 7) | ((kr >> 4) << 3);
        int mt = m >> 4;
        As[(((mt << 3) + kk) * 32 + lane_) * 16 + e] =
            xedw[((size_t)m * NN + n) * 256 + kcol];
    }
    __syncthreads();

    int wave = threadIdx.x >> 5, lane = threadIdx.x & 31;
    int mt = wave & 1, nt = wave >> 1;
    int lhi = lane >> 4, l15 = lane & 15;

    v8f acc = {};
    #pragma unroll
    for (int kk = 0; kk < 8; ++kk) {
        v16bf a   = *(const v16bf*)&As[(((mt << 3) + kk) * 32 + lane) * 16];
        v16bf bfr = *(const v16bf*)&Bs[(((nt << 3) + kk) * 32 + lane) * 16];
        acc = __builtin_amdgcn_wmma_f32_16x16x32_bf16(false, a, false, bfr,
                                                      (short)0, acc, false, false);
    }

    // D store: f32 x1 + bf16 x1swz (B-swizzled for the adj kernel: m-dim = n)
    int kk6 = nt >> 1;                         // (nt*16+l15) >> 5
    int lane6 = (nt & 1) * 16 + (n & 15);
    size_t swz_in_b = ((size_t)((n >> 4) * 2 + kk6) * 32 + lane6) * 16 + l15;
    #pragma unroll
    for (int r = 0; r < 8; ++r) {
        int brow = mt * 16 + r + (lhi << 3);
        int e    = nt * 16 + l15;
        float v  = fmaxf(acc[r], 0.0f);
        x1[((size_t)brow * NN + n) * ED + e] = v;
        x1swz[(size_t)brow * 65536 + swz_in_b] = (__bf16)v;
    }
}

// =====================================================================
// Kernel 5: LayerNorm stats over (N,ED) per batch
// =====================================================================
__global__ void k_lnstats(const float* __restrict__ x1, float* __restrict__ stats) {
    int b = blockIdx.x;
    __shared__ float s1[1024], s2[1024];
    const float* row = x1 + ((size_t)b * NN + threadIdx.x) * ED;
    float a = 0.0f, q = 0.0f;
    #pragma unroll 8
    for (int e = 0; e < ED; ++e) { float v = row[e]; a += v; q += v * v; }
    s1[threadIdx.x] = a; s2[threadIdx.x] = q;
    __syncthreads();
    for (int w = 512; w; w >>= 1) {
        if (threadIdx.x < w) { s1[threadIdx.x] += s1[threadIdx.x + w];
                               s2[threadIdx.x] += s2[threadIdx.x + w]; }
        __syncthreads();
    }
    if (threadIdx.x == 0) {
        float mu  = s1[0] * (1.0f / 65536.0f);
        float var = s2[0] * (1.0f / 65536.0f) - mu * mu;
        stats[b * 2]     = mu;
        stats[b * 2 + 1] = rsqrtf(var + 1e-8f);
    }
}

// =====================================================================
// Kernel 6 (fused): DEw = LN(x1)@Wxabs, adj = relu(DEw @ x1^T),
//                   top-30 mask, softmax -> out.  adj never hits HBM.
// x1 B-operand (128 KB, pre-swizzled in global) arrives via
// global_load_async_to_lds_b128 (ASYNCcnt), overlapped with prep.
// Dynamic LDS = 204 KB (< 320 KB/WGP).
// =====================================================================
#define L_ADJ   0
#define L_X1B   65536          // 64*2*32*16*2 = 131072
#define L_X1K   196608         // 2*32*16*2    = 2048
#define L_DEW   198656         // 2048
#define L_WXA   200704         // 4*2*32*16*2  = 8192
#define L_TOTAL 208896

__global__ void __launch_bounds__(256) k_adj_topk_softmax(
        const float* __restrict__ x1, const __bf16* __restrict__ x1swz,
        const float* __restrict__ Wxabs, const float* __restrict__ stats,
        float* __restrict__ out) {
    extern __shared__ __align__(32) char smem[];
    float*  adjr = (float*)(smem + L_ADJ);     // [16][1024]
    __bf16* x1b  = (__bf16*)(smem + L_X1B);    // [mt 64][kk 2][lane 32][e 16]
    __bf16* x1k  = (__bf16*)(smem + L_X1K);    // [kk 2][lane 32][e 16]
    __bf16* dew  = (__bf16*)(smem + L_DEW);    // [kk 2][lane 32][e 16]
    __bf16* wxa  = (__bf16*)(smem + L_WXA);    // [nt 4][kk 2][lane 32][e 16]

    int b  = blockIdx.x >> 6;
    int n0 = (blockIdx.x & 63) << 4;
    int tid = threadIdx.x;
    int wave = tid >> 5, lane = tid & 31;
    int lhi = lane >> 4, l15 = lane & 15;

    float mu = stats[b * 2], rsig = stats[b * 2 + 1];
    const float* xb = x1 + (size_t)b * NN * ED;

    // ---- async DMA: pre-swizzled x1 bf16 (128 KB contiguous) -> LDS ----
    {
        unsigned lbase = (unsigned)(uintptr_t)(void*)x1b;   // LDS aperture: low 32b
        const char* g = (const char*)(x1swz + (size_t)b * 65536);
        #pragma unroll 4
        for (int it = 0; it < 32; ++it) {
            int idx = it * 256 + tid;
            unsigned loff = lbase + idx * 16;
            unsigned long long ga = (unsigned long long)(uintptr_t)(g + idx * 16);
            asm volatile("global_load_async_to_lds_b128 %0, %1, off"
                         :: "v"(loff), "v"(ga) : "memory");
        }
    }

    // ---- overlap: LN'd A rows (A-swizzled) + Wxabs (B-swizzled) ----
    for (int i = tid; i < 16 * ED; i += 256) {
        int r = i >> 6, k = i & 63;
        float v = xb[(size_t)(n0 + r) * ED + k];
        int kk = k >> 5, kr = k & 31;
        int lane_ = ((kr >> 3) & 1) * 16 + r;
        int e = (kr & 7) | ((kr >> 4) << 3);
        x1k[(kk * 32 + lane_) * 16 + e] = (__bf16)((v - mu) * rsig);
    }
    for (int i = tid; i < ED * ED; i += 256) {
        int k = i >> 6, col = i & 63;
        int kk = k >> 5, kr = k & 31;
        int lane_ = ((kr >> 3) & 1) * 16 + (col & 15);
        int e = (kr & 7) | ((kr >> 4) << 3), nt = col >> 4;
        wxa[((nt * 2 + kk) * 32 + lane_) * 16 + e] = (__bf16)Wxabs[i];
    }
    asm volatile("s_wait_asynccnt 0x0" ::: "memory");
    __syncthreads();

    // ---- DEw tile [16,64] = x1k @ Wxabs (waves 0..3), stored A-swizzled ----
    if (wave < 4) {
        int nt = wave;
        v8f acc = {};
        #pragma unroll
        for (int kk = 0; kk < 2; ++kk) {
            v16bf a   = *(const v16bf*)&x1k[(kk * 32 + lane) * 16];
            v16bf bfr = *(const v16bf*)&wxa[((nt * 2 + kk) * 32 + lane) * 16];
            acc = __builtin_amdgcn_wmma_f32_16x16x32_bf16(false, a, false, bfr,
                                                          (short)0, acc, false, false);
        }
        #pragma unroll
        for (int r = 0; r < 8; ++r) {
            int srow = r + (lhi << 3);           // n-local row (A-matrix m)
            int scol = nt * 16 + l15;            // k
            int kk2 = scol >> 5, kr2 = scol & 31;
            int lane2 = (kr2 >> 4) * 16 + srow;
            int e2 = kr2 & 15;
            dew[(kk2 * 32 + lane2) * 16 + e2] = (__bf16)acc[r];
        }
    }
    __syncthreads();

    // ---- adj slab [16,1024]: A-frags invariant across m-tiles (hoisted) ----
    v16bf a0 = *(const v16bf*)&dew[(0 * 32 + lane) * 16];
    v16bf a1 = *(const v16bf*)&dew[(1 * 32 + lane) * 16];
    #pragma unroll
    for (int t8 = 0; t8 < 8; ++t8) {
        int mt = wave * 8 + t8;
        v8f acc = {};
        v16bf b0 = *(const v16bf*)&x1b[((mt * 2 + 0) * 32 + lane) * 16];
        acc = __builtin_amdgcn_wmma_f32_16x16x32_bf16(false, a0, false, b0,
                                                      (short)0, acc, false, false);
        v16bf b1 = *(const v16bf*)&x1b[((mt * 2 + 1) * 32 + lane) * 16];
        acc = __builtin_amdgcn_wmma_f32_16x16x32_bf16(false, a1, false, b1,
                                                      (short)0, acc, false, false);
        #pragma unroll
        for (int r = 0; r < 8; ++r)
            adjr[(r + (lhi << 3)) * 1024 + mt * 16 + l15] = fmaxf(acc[r], 0.0f);
    }
    __syncthreads();

    // ---- per-row top-30 + masked softmax (2 rows per wave) ----
    for (int rr = 0; rr < 2; ++rr) {
        int row = wave * 2 + rr;
        float vals[32];
        #pragma unroll
        for (int i = 0; i < 32; ++i) vals[i] = adjr[row * 1024 + i * 32 + lane];

        unsigned sel = 0u, exc = 0u;
        float rowmax = 0.0f;
        for (int j = 0; j < TOPK; ++j) {
            float bv = -1.0f; int bi = 1 << 30;
            #pragma unroll
            for (int i = 0; i < 32; ++i) {
                float v = ((exc >> i) & 1u) ? -1.0f : vals[i];
                int gi = i * 32 + lane;
                if (v > bv || (v == bv && gi < bi)) { bv = v; bi = gi; }
            }
            #pragma unroll
            for (int off = 16; off; off >>= 1) {
                float ov = __shfl_xor(bv, off, 32);
                int   oi = __shfl_xor(bi, off, 32);
                if (ov > bv || (ov == bv && oi < bi)) { bv = ov; bi = oi; }
            }
            if (j == 0) rowmax = bv;                    // relu => >= 0
            if ((bi & 31) == lane) { unsigned ii = (unsigned)bi >> 5;
                                     exc |= 1u << ii; sel |= 1u << ii; }
        }

        float s = 0.0f;
        #pragma unroll
        for (int i = 0; i < 32; ++i)
            if ((sel >> i) & 1u) s += __expf(vals[i] - rowmax);
        #pragma unroll
        for (int off = 16; off; off >>= 1) s += __shfl_xor(s, off, 32);

        float e0  = __expf(-rowmax);                    // masked entries == 0
        float inv = 1.0f / (s + (float)(NN - TOPK) * e0);
        float* outp = out + ((size_t)b * NN + (n0 + row)) * NN;
        #pragma unroll
        for (int i = 0; i < 32; ++i) {
            float o = ((sel >> i) & 1u) ? __expf(vals[i] - rowmax) * inv : e0 * inv;
            outp[i * 32 + lane] = o;
        }
    }
}

// =====================================================================
extern "C" void kernel_launch(void* const* d_in, const int* in_sizes, int n_in,
                              void* d_out, int out_size, void* d_ws, size_t ws_size,
                              hipStream_t stream) {
    const float* x     = (const float*)d_in[0];
    const float* T_D   = (const float*)d_in[1];
    const float* D_W   = (const float*)d_in[2];
    const float* E     = (const float*)d_in[3];
    const float* Wx    = (const float*)d_in[4];
    const float* Wd    = (const float*)d_in[5];
    const float* Wxabs = (const float*)d_in[6];
    float* out = (float*)d_out;

    char* ws = (char*)d_ws;
    float*  xn1   = (float*)(ws + OFF_XN1);
    __bf16* xedw  = (__bf16*)(ws + OFF_XEDW);
    float*  x1    = (float*)(ws + OFF_X1);
    float*  stats = (float*)(ws + OFF_STATS);
    __bf16* x1swz = (__bf16*)(ws + OFF_X1SWZ);

    k_dft    <<<(BB * NN) / 128, 128, 0, stream>>>(x, xn1);
    k_norm2  <<<BB * FFT_DIM,    256, 0, stream>>>(xn1);
    k_xedw   <<<BB * (NN / 4),   256, 0, stream>>>(xn1, E, T_D, D_W, Wx, xedw);
    k_node_mm<<<NN,              256, 0, stream>>>(xedw, Wd, x1, x1swz);
    k_lnstats<<<BB,             1024, 0, stream>>>(x1, stats);

    k_adj_topk_softmax<<<BB * (NN / 16), 256, (size_t)L_TOTAL, stream>>>(
        x1, x1swz, Wxabs, stats, out);

    (void)in_sizes; (void)n_in; (void)out_size; (void)ws_size;
}
